// GraphRecommendationModel_19258633355264
// MI455X (gfx1250) — matrix operand order repaired and checked by reading, not compiled, not run on hardware
//
#include <hip/hip_runtime.h>
#include <hip/hip_bf16.h>
#include <math.h>

#if defined(__HIP_DEVICE_COMPILE__) && __has_builtin(__builtin_amdgcn_global_load_async_to_lds_b128)
#define HAS_ASYNC_LDS 1
#else
#define HAS_ASYNC_LDS 0
#endif

typedef __attribute__((ext_vector_type(16))) _Float16 v16h;
typedef __attribute__((ext_vector_type(8)))  _Float16 v8h;
typedef __attribute__((ext_vector_type(4)))  _Float16 v4h;
typedef __attribute__((ext_vector_type(8)))  float    v8f;
typedef __attribute__((ext_vector_type(4)))  float    v4f;

#define BDIM 256
#define LDIM 50
#define NN   16
#define HH   128

// LDS row strides (elements); all multiples of 8 halfs (16B) for b128 frag loads
#define WAT_S 264   // WaT  [96][264]  f16 (col-major Wa: row n holds K=0..255)
#define W1T_S 104   // W1T  [48][104]  f16
#define W2T_S 72    // W2T  [16][72]   f16 (rows K=48..63 zero-padded)
#define ATT_S 264   // att  [16][264]  f16 (A-tile: hist | neighbor, K=0..255)
#define SC_S  104   // sc   [16][104]  f16 (scores, K=0..95)
#define H1_S  72    // h1   [16][72]   f16 (cols 48..63 zero)
#define NB_S  132   // nb   [16][132]  f32 (neighbor copy, double buffered)

// byte offsets of the f16 region
#define OFF_WAT   0
#define OFF_W1T   (OFF_WAT + 96*WAT_S*2)
#define OFF_W2T   (OFF_W1T + 48*W1T_S*2)
#define OFF_ATT   (OFF_W2T + 16*W2T_S*2)
#define OFF_SC    (OFF_ATT + 16*ATT_S*2)
#define OFF_H1    (OFF_SC  + 16*SC_S*2)
#define OFF_HIST  (OFF_H1  + 16*H1_S*2)
#define OFF_F32   (OFF_HIST + 128*2)
#define NF32 (2*16*NB_S + 256 + 96 + 48 + 16 + 16 + 16 + 16 + 16 + 128 + 128 + 128 + 128)
#define SMEM_BYTES (OFF_F32 + NF32*4)

__device__ __forceinline__ v16h frag16(const _Float16* p0, const _Float16* p1) {
    v8h lo = *(const v8h*)p0;
    v8h hi = *(const v8h*)p1;
    return __builtin_shufflevector(lo, hi, 0,1,2,3,4,5,6,7,8,9,10,11,12,13,14,15);
}

// A-frag: per-lane halfs = K in {k0..k0+7} ++ {k0+16..k0+23}, k0 = 32*ks + 8*(lane>=16)
// B-frag: per-lane halfs = K in {k0..k0+7} ++ {k0+16..k0+23}, k0 = 32*ks + 8*(lane>=16)
template<int KSTEPS>
__device__ __forceinline__ v8f gemm_tile(const _Float16* A, int as,
                                         const _Float16* Brow, int lane) {
    v8f c = {};
    const _Float16* arow = A + (lane & 15) * as;
    const int aoff = (lane & 16) ? 8 : 0;
    const int boff = (lane & 16) ? 8 : 0;
#pragma unroll
    for (int ks = 0; ks < KSTEPS; ++ks) {
        const _Float16* ap = arow + ks * 32 + aoff;
        const _Float16* bp = Brow + ks * 32 + boff;
        v16h af = frag16(ap, ap + 16);
        v16h bf = frag16(bp, bp + 16);
        c = __builtin_amdgcn_wmma_f32_16x16x32_f16(
                false, af, false, bf, (short)0, c, false, false);
    }
    return c;
}

#if HAS_ASYNC_LDS
// builtin parameter types: (AS1 int4*, AS3 int4*, imm offset, imm cpol)
typedef int v4i_gcc __attribute__((vector_size(16)));
typedef __attribute__((address_space(1))) v4i_gcc gv4i;
typedef __attribute__((address_space(3))) v4i_gcc lv4i;

__device__ __forceinline__ void waitAsync2() {
#if __has_builtin(__builtin_amdgcn_s_wait_asynccnt)
    __builtin_amdgcn_s_wait_asynccnt(2);
#else
    asm volatile("s_wait_asynccnt 0x2" ::: "memory");
#endif
}
__device__ __forceinline__ void waitAsync0() {
#if __has_builtin(__builtin_amdgcn_s_wait_asynccnt)
    __builtin_amdgcn_s_wait_asynccnt(0);
#else
    asm volatile("s_wait_asynccnt 0x0" ::: "memory");
#endif
}
// 8KB tile = 512 float4 chunks; 256 threads x 2 chunks, non-temporal (streamed once)
__device__ __forceinline__ void issue_nb_async(const float* nbg, float* nbdst, int tid) {
#pragma unroll
    for (int r = 0; r < 2; ++r) {
        int j = tid + r * BDIM;          // float4 chunk index 0..511
        int m = j >> 5;                  // neighbor row
        int k = (j & 31) * 4;            // column within row
        __builtin_amdgcn_global_load_async_to_lds_b128(
            (gv4i*)(nbg + (size_t)j * 4),
            (lv4i*)(nbdst + m * NB_S + k),
            0, /*cpol: TH=NT*/ 1);
    }
}
#endif

__global__ void __launch_bounds__(BDIM)
rec_fused_kernel(const int* __restrict__ user_ids,
                 const int* __restrict__ history,
                 const float* __restrict__ neighbor_emb,
                 const float* __restrict__ user_table,
                 const float* __restrict__ item_table,
                 const float* __restrict__ Wa,  const float* __restrict__ ba,
                 const float* __restrict__ W1,  const float* __restrict__ b1,
                 const float* __restrict__ W2,  const float* __restrict__ b2,
                 const float* __restrict__ W3,  const float* __restrict__ b3,
                 const float* __restrict__ Wmh, const float* __restrict__ bmh,
                 const float* __restrict__ Wfc1,const float* __restrict__ bfc1,
                 const float* __restrict__ Wout,const float* __restrict__ bout,
                 float* __restrict__ out)
{
    extern __shared__ char smem[];
    _Float16* WaT   = (_Float16*)(smem + OFF_WAT);
    _Float16* W1T   = (_Float16*)(smem + OFF_W1T);
    _Float16* W2T   = (_Float16*)(smem + OFF_W2T);
    _Float16* att   = (_Float16*)(smem + OFF_ATT);
    _Float16* sc    = (_Float16*)(smem + OFF_SC);
    _Float16* h1    = (_Float16*)(smem + OFF_H1);
    _Float16* histh = (_Float16*)(smem + OFF_HIST);
    float* nb0    = (float*)(smem + OFF_F32);   // [16][132]
    float* nb1    = nb0  + 16*NB_S;             // [16][132]
    float* h2f    = nb1  + 16*NB_S;             // [16][16]
    float* baL    = h2f  + 256;                 // 96
    float* b1L    = baL  + 96;                  // 48
    float* b2L    = b1L  + 48;                  // 16
    float* w3L    = b2L  + 16;                  // 16
    float* lgL    = w3L  + 16;                  // 16
    float* eeL    = lgL  + 16;                  // 16
    float* attnL  = eeL  + 16;                  // 16
    float* aggs   = attnL+ 16;                  // 128
    float* pooled = aggs + 128;                 // 128
    float* us     = pooled + 128;               // 128
    float* hidL   = us   + 128;                 // 128

    const int b    = blockIdx.x;
    const int tid  = threadIdx.x;
    const int lane = tid & 31;
    const int wave = tid >> 5;

    const float* nbase = neighbor_emb + (size_t)b * LDIM * NN * HH;

#if HAS_ASYNC_LDS
    issue_nb_async(nbase, nb0, tid);              // prefetch l=0 under weight prep
#endif

    // history-embedding software pipeline (register prefetch)
    float hvcur = 0.f;
    if (tid < HH) hvcur = item_table[(size_t)history[b*LDIM]*HH + tid];

    // ---- zero pads first (ordered before fills) ----
    for (int i = tid; i < 16*W2T_S; i += BDIM) W2T[i] = (_Float16)0.f;
    for (int i = tid; i < 16*H1_S;  i += BDIM) h1[i]  = (_Float16)0.f;
    __syncthreads();

    // ---- per-block weight prep: f16 transposed copies in LDS ----
    for (int idx = tid; idx < 256*96; idx += BDIM) {       // Wa [256][96] -> WaT[n][k]
        int n = idx % 96, k = idx / 96;
        WaT[n*WAT_S + k] = (_Float16)Wa[idx];
    }
    for (int idx = tid; idx < 96*48; idx += BDIM) {        // W1 [96][48] -> W1T[n][k]
        int n = idx % 48, k = idx / 48;
        W1T[n*W1T_S + k] = (_Float16)W1[idx];
    }
    for (int idx = tid; idx < 48*16; idx += BDIM) {        // W2 [48][16] -> W2T[n][k]
        int n = idx % 16, k = idx / 16;
        W2T[n*W2T_S + k] = (_Float16)W2[idx];
    }
    if (tid < 96)  baL[tid] = ba[tid];
    if (tid < 48)  b1L[tid] = b1[tid];
    if (tid < 16)  b2L[tid] = b2[tid];
    if (tid < 16)  w3L[tid] = W3[tid];
    if (tid < HH) {
        aggs[tid] = 0.f;
        us[tid]   = user_table[(size_t)user_ids[b]*HH + tid];
    }
    __syncthreads();

    const float b3s = b3[0];

    for (int l = 0; l < LDIM; ++l) {
        float* nbcur  = (l & 1) ? nb1 : nb0;
#if HAS_ASYNC_LDS
        float* nbnext = (l & 1) ? nb0 : nb1;
#endif
        // publish current history column; prefetch next (overlaps whole step)
        if (tid < HH) histh[tid] = (_Float16)hvcur;
        float hvnext = 0.f;
        if (l + 1 < LDIM) {
#if HAS_ASYNC_LDS
            issue_nb_async(nbase + (size_t)(l + 1) * NN * HH, nbnext, tid);
#endif
            if (tid < HH)
                hvnext = item_table[(size_t)history[b*LDIM + l + 1]*HH + tid];
        }
#if HAS_ASYNC_LDS
        if (l + 1 < LDIM) waitAsync2(); else waitAsync0();
#endif
        __syncthreads();   // nbcur complete + histh visible

        // ---- build A-tile [16 x 256] f16 (hist cols 0..127 | neighbor 128..255) ----
#pragma unroll
        for (int r = 0; r < 2; ++r) {
            int j = tid + r * BDIM;      // float4 chunk 0..511
            int m = j >> 5;
            int k = (j & 31) * 4;
#if HAS_ASYNC_LDS
            v4f v = *(const v4f*)(nbcur + m*NB_S + k);
#else
            v4f v = *(const v4f*)(nbase + (size_t)l*NN*HH + (size_t)j*4);
            *(v4f*)(nbcur + m*NB_S + k) = v;
#endif
            *(v4h*)(att + m*ATT_S + HH + k) = __builtin_convertvector(v, v4h);
            *(v4h*)(att + m*ATT_S + k)      = *(const v4h*)(histh + k);
        }
        __syncthreads();

        // ---- GEMM1: [16x256]x[256x96] -> scores, relu, f16 (waves 0..5) ----
        if (wave < 6) {
            const _Float16* brow = WaT + (wave*16 + (lane & 15)) * WAT_S;
            v8f c = gemm_tile<8>(att, ATT_S, brow, lane);
            const int col = wave*16 + (lane & 15);
            const int mb  = (lane & 16) ? 8 : 0;
            const float bias = baL[col];
#pragma unroll
            for (int i = 0; i < 8; ++i)
                sc[(mb + i)*SC_S + col] = (_Float16)fmaxf(c[i] + bias, 0.f);
        }
        __syncthreads();

        // ---- GEMM2: [16x96]x[96x48] -> h1, relu, f16 (waves 0..2) ----
        if (wave < 3) {
            const _Float16* brow = W1T + (wave*16 + (lane & 15)) * W1T_S;
            v8f c = gemm_tile<3>(sc, SC_S, brow, lane);
            const int col = wave*16 + (lane & 15);
            const int mb  = (lane & 16) ? 8 : 0;
            const float bias = b1L[col];
#pragma unroll
            for (int i = 0; i < 8; ++i)
                h1[(mb + i)*H1_S + col] = (_Float16)fmaxf(c[i] + bias, 0.f);
        }
        __syncthreads();

        // ---- GEMM3: [16x64(pad)]x[64x16] -> h2, relu, f32 (wave 0) ----
        if (wave == 0) {
            const _Float16* brow = W2T + (lane & 15) * W2T_S;
            v8f c = gemm_tile<2>(h1, H1_S, brow, lane);
            const int col = lane & 15;
            const int mb  = (lane & 16) ? 8 : 0;
            const float bias = b2L[col];
#pragma unroll
            for (int i = 0; i < 8; ++i)
                h2f[(mb + i)*16 + col] = fmaxf(c[i] + bias, 0.f);
        }
        __syncthreads();

        // ---- logits + softmax over 16 neighbors (f32, exact) ----
        if (tid < NN) {
            float s = b3s;
#pragma unroll
            for (int k = 0; k < 16; ++k) s += h2f[tid*16 + k] * w3L[k];
            lgL[tid] = s;
        }
        __syncthreads();
        if (tid < NN) {
            float mx = lgL[0];
#pragma unroll
            for (int k = 1; k < 16; ++k) mx = fmaxf(mx, lgL[k]);
            eeL[tid] = expf(lgL[tid] - mx);
        }
        __syncthreads();
        if (tid < NN) {
            float s = 0.f;
#pragma unroll
            for (int k = 0; k < 16; ++k) s += eeL[k];
            attnL[tid] = eeL[tid] / s;
        }
        __syncthreads();

        // ---- weighted aggregation, accumulated over l (Wmh is linear) ----
        if (tid < HH) {
            float s = 0.f;
#pragma unroll
            for (int n = 0; n < NN; ++n) s += attnL[n] * nbcur[n*NB_S + tid];
            aggs[tid] += s;
        }
        hvcur = hvnext;
        __syncthreads();   // protect nbcur/att/sc/h1/histh before next iteration
    }

    // ---- pooled = (mean_l agg) @ Wmh + bmh ----
    if (tid < HH) aggs[tid] *= (1.f / (float)LDIM);
    __syncthreads();
    if (tid < HH) {
        float v = bmh[tid];
        for (int k = 0; k < HH; ++k) v += aggs[k] * Wmh[k*HH + tid];
        pooled[tid] = v;
    }
    __syncthreads();

    // ---- hidden = relu([user|pooled] @ Wfc1 + bfc1) ----
    if (tid < HH) {
        float v = bfc1[tid];
        for (int k = 0; k < HH; ++k) v += us[k]     * Wfc1[k*HH + tid];
        for (int k = 0; k < HH; ++k) v += pooled[k] * Wfc1[(HH + k)*HH + tid];
        hidL[tid] = fmaxf(v, 0.f);
    }
    __syncthreads();

    // ---- out = sigmoid(hidden @ Wout + bout) ----
    if (tid == 0) {
        float s = bout[0];
        for (int k = 0; k < HH; ++k) s += hidL[k] * Wout[k];
        out[b] = 1.f / (1.f + expf(-s));
    }
}

extern "C" void kernel_launch(void* const* d_in, const int* in_sizes, int n_in,
                              void* d_out, int out_size, void* d_ws, size_t ws_size,
                              hipStream_t stream) {
    const int*   user_ids = (const int*)  d_in[0];
    // d_in[1] item_ids: unused in reference
    const int*   history  = (const int*)  d_in[2];
    const float* neighbor = (const float*)d_in[3];
    const float* user_tab = (const float*)d_in[4];
    const float* item_tab = (const float*)d_in[5];
    const float* Wa   = (const float*)d_in[6];
    const float* ba   = (const float*)d_in[7];
    const float* W1   = (const float*)d_in[8];
    const float* b1   = (const float*)d_in[9];
    const float* W2   = (const float*)d_in[10];
    const float* b2   = (const float*)d_in[11];
    const float* W3   = (const float*)d_in[12];
    const float* b3   = (const float*)d_in[13];
    const float* Wmh  = (const float*)d_in[14];
    const float* bmh  = (const float*)d_in[15];
    const float* Wfc1 = (const float*)d_in[16];
    const float* bfc1 = (const float*)d_in[17];
    const float* Wout = (const float*)d_in[18];
    const float* bout = (const float*)d_in[19];
    float* out = (float*)d_out;

    const int B = in_sizes[0];   // 512

    (void)hipFuncSetAttribute((const void*)rec_fused_kernel,
                              hipFuncAttributeMaxDynamicSharedMemorySize,
                              (int)SMEM_BYTES);

    rec_fused_kernel<<<B, BDIM, SMEM_BYTES, stream>>>(
        user_ids, history, neighbor, user_tab, item_tab,
        Wa, ba, W1, b1, W2, b2, W3, b3, Wmh, bmh, Wfc1, bfc1, Wout, bout, out);
}
